// GCNN_2L_4982162063670
// MI455X (gfx1250) — matrix operand order repaired and checked by reading, not compiled, not run on hardware
//
#include <hip/hip_runtime.h>

// ---------------------------------------------------------------------------
// 2-layer GCN (DGL GraphConv, norm='both') for MI455X / gfx1250.
// GEMMs: v_wmma_f32_16x16x32_bf16 (bf16 in, f32 accumulate), double-buffered
// LDS fed by global_load_async_to_lds_b128 (ASYNCcnt-tracked, no VGPR bounce).
// Aggregation = edge scatter-add with f32 global atomics (L2-resident).
// Scratch layout (ws, ~404 MB):
//   [inv_dout: N f32][inv_din: N f32][W1T bf16 512x1024][W2T bf16 256x512]
//   [Hs bf16 N*1024 | reused as H1s bf16 N*512]
//   [X1 f32 N*512   | reused as X2 f32 N*256]
//   [AGG1 f32 N*512]
// ---------------------------------------------------------------------------

typedef __attribute__((ext_vector_type(16))) __bf16 v16bf;
typedef __attribute__((ext_vector_type(8)))  __bf16 v8bf;
typedef __attribute__((ext_vector_type(4)))  __bf16 v4bf;
typedef __attribute__((ext_vector_type(8)))  float  v8f;

#define IN_DIM  1024
#define HID_DIM 512
#define OUT_DIM 256

#define USE_ASYNC_LDS 1

union V16 { v16bf v; v8bf h[2]; };

// 16-byte global -> LDS copy. Async path: per-lane DMA into LDS, tracked by
// ASYNCcnt; VGPR operand is the 32-bit LDS offset (low 32 bits of the flat
// shared-memory address == AS(3) offset).
__device__ __forceinline__ void copy16_g2s(const __bf16* __restrict__ g, __bf16* s) {
#if USE_ASYNC_LDS
  unsigned loff = (unsigned)(unsigned long long)s;
  asm volatile("global_load_async_to_lds_b128 %0, %1, off"
               :: "v"(loff), "v"(g) : "memory");
#else
  *(v8bf*)s = *(const v8bf*)g;
#endif
}

__device__ __forceinline__ void wait_async_tail() {   // all but newest 5 done
#if USE_ASYNC_LDS
  asm volatile("s_wait_asynccnt 5" ::: "memory");
#endif
}
__device__ __forceinline__ void wait_async_all() {
#if USE_ASYNC_LDS
  asm volatile("s_wait_asynccnt 0" ::: "memory");
#endif
}

// ---------------- degree / normalization ----------------
__global__ void k_deg(const long long* __restrict__ src, const long long* __restrict__ dst,
                      float* __restrict__ deg_out, float* __restrict__ deg_in, int E) {
  int e = blockIdx.x * blockDim.x + threadIdx.x;
  if (e < E) {
    atomicAdd(&deg_out[(int)src[e]], 1.0f);
    atomicAdd(&deg_in [(int)dst[e]], 1.0f);
  }
}

__global__ void k_rsqrt(float* __restrict__ p, int n) {
  int i = blockIdx.x * blockDim.x + threadIdx.x;
  if (i < n) p[i] = rsqrtf(fmaxf(p[i], 1.0f));
}

// ---------------- X[i,:] * inv[i] -> bf16 (8 elems / thread) ----------------
__global__ void k_scale_cvt(const float* __restrict__ X, const float* __restrict__ inv,
                            __bf16* __restrict__ Y, int F, long total8) {
  long idx = blockIdx.x * (long)blockDim.x + threadIdx.x;
  if (idx >= total8) return;
  int row = (int)(idx / (F / 8));
  int c   = (int)(idx % (F / 8)) * 8;
  const float4* p = (const float4*)&X[(long)row * F + c];
  float4 x0 = p[0], x1 = p[1];
  float s = inv[row];
  v8bf y;
  y[0] = (__bf16)(x0.x * s); y[1] = (__bf16)(x0.y * s);
  y[2] = (__bf16)(x0.z * s); y[3] = (__bf16)(x0.w * s);
  y[4] = (__bf16)(x1.x * s); y[5] = (__bf16)(x1.y * s);
  y[6] = (__bf16)(x1.z * s); y[7] = (__bf16)(x1.w * s);
  *(v8bf*)&Y[(long)row * F + c] = y;
}

// ---------------- W [K,N] f32 -> W^T [N,K] bf16 ----------------
__global__ void k_cvt_T(const float* __restrict__ W, __bf16* __restrict__ WT, int K, int Nn) {
  int idx = blockIdx.x * blockDim.x + threadIdx.x;
  if (idx >= K * Nn) return;
  int n = idx / K, k = idx % K;   // consecutive threads -> consecutive k: coalesced writes
  WT[(long)n * K + k] = (__bf16)W[(long)k * Nn + n];
}

// ---------------- bf16 WMMA GEMM: C[M,Nn] = A[M,K] @ BT[Nn,K]^T ----------------
// Block: 256 threads = 8 waves (2x4). Block tile 64(M) x 256(N), K-step 32.
// Double-buffered LDS filled by async-to-LDS; wave tile 32x64 = 2x4 WMMA accs.
__global__ __launch_bounds__(256) void gemm_bf16_wmma(
    const __bf16* __restrict__ A,   // [M,K] row-major (bf16)
    const __bf16* __restrict__ BT,  // [Nn,K] row-major (B transposed, bf16)
    float* __restrict__ C,          // [M,Nn] f32
    int M, int Nn, int K)
{
  const int SAS = 40, SBS = 40;     // padded LDS strides (multiple of 8 -> 16B aligned)
  __shared__ __align__(16) __bf16 sA[2][64 * 40];
  __shared__ __align__(16) __bf16 sB[2][256 * 40];

  const int tid  = threadIdx.x;
  const int lane = tid & 31;
  const int wid  = tid >> 5;
  const int wm   = wid >> 2;        // 0..1 : wave row  (32 rows each)
  const int wn   = wid & 3;         // 0..3 : wave col  (64 cols each)
  const int half = lane >> 4;       // lane group 0-15 / 16-31
  const int lrow = lane & 15;

  const int m0 = blockIdx.y * 64;
  const int n0 = blockIdx.x * 256;

  const v8f vzero = {0.f, 0.f, 0.f, 0.f, 0.f, 0.f, 0.f, 0.f};
  v8f acc[2][4];
  #pragma unroll
  for (int i = 0; i < 2; i++)
    #pragma unroll
    for (int j = 0; j < 4; j++) acc[i][j] = vzero;

  const int arow = tid >> 2;        // 0..63
  const int acg  = (tid & 3) * 8;   // 0/8/16/24

  // stage tile (A 64x32, BT 256x32) for K-offset kt into buffer `buf`
  auto issue = [&](int kt, int buf) {
    copy16_g2s(&A[(long)(m0 + arow) * K + kt + acg], &sA[buf][arow * SAS + acg]);
    #pragma unroll
    for (int r = 0; r < 4; r++) {
      int brow = r * 64 + arow;
      copy16_g2s(&BT[(long)(n0 + brow) * K + kt + acg], &sB[buf][brow * SBS + acg]);
    }
  };

  const int nk = K / 32;
  issue(0, 0);

  for (int i = 0; i < nk; ++i) {
    const int buf = i & 1;
    if (i + 1 < nk) {             // prefetch next tile into the other buffer
      issue((i + 1) * 32, buf ^ 1);
      wait_async_tail();          // tile i landed; tile i+1 (5 ops) in flight
    } else {
      wait_async_all();
    }
    __syncthreads();              // tile i visible to all waves

    // A fragment: row = lrow; j<8 -> K = half*8+j ; j>=8 -> K = 16+half*8+(j-8)
    V16 a[2], b[4];
    #pragma unroll
    for (int mi = 0; mi < 2; mi++) {
      int row = wm * 32 + mi * 16 + lrow;
      a[mi].h[0] = *(const v8bf*)&sA[buf][row * SAS + half * 8];
      a[mi].h[1] = *(const v8bf*)&sA[buf][row * SAS + 16 + half * 8];
    }
    // B fragment from BT rows: col = lrow; j<8 -> K = half*8+j ; j>=8 -> K = 16+half*8+(j-8)
    #pragma unroll
    for (int ni = 0; ni < 4; ni++) {
      int row = wn * 64 + ni * 16 + lrow;
      b[ni].h[0] = *(const v8bf*)&sB[buf][row * SBS + half * 8];
      b[ni].h[1] = *(const v8bf*)&sB[buf][row * SBS + 16 + half * 8];
    }

    #pragma unroll
    for (int mi = 0; mi < 2; mi++)
      #pragma unroll
      for (int ni = 0; ni < 4; ni++)
        acc[mi][ni] = __builtin_amdgcn_wmma_f32_16x16x32_bf16(
            false, a[mi].v, false, b[ni].v, (short)0, acc[mi][ni], false, false);

    __syncthreads();              // all waves done reading buf before it is refilled
  }

  // C layout: VGPR r -> M = half*8 + r, N = lrow
  #pragma unroll
  for (int mi = 0; mi < 2; mi++)
    #pragma unroll
    for (int ni = 0; ni < 4; ni++) {
      int col = n0 + wn * 64 + ni * 16 + lrow;
      #pragma unroll
      for (int r = 0; r < 8; r++) {
        int row = m0 + wm * 32 + mi * 16 + half * 8 + r;
        C[(long)row * Nn + col] = acc[mi][ni][r];
      }
    }
}

// ---------------- edge scatter-add: AGG[dst] += X[src], 4 floats / thread ----------------
__global__ void k_scatter4(const long long* __restrict__ src, const long long* __restrict__ dst,
                           const float* __restrict__ X, float* __restrict__ AGG,
                           int F4, long total) {
  long idx = blockIdx.x * (long)blockDim.x + threadIdx.x;
  if (idx >= total) return;
  int e = (int)(idx / F4);
  int c = (int)(idx % F4) * 4;
  long F = (long)F4 * 4;
  long s = (long)src[e], d = (long)dst[e];
  const float4 v = *(const float4*)&X[s * F + c];
  float* o = &AGG[d * F + c];
  atomicAdd(o + 0, v.x);
  atomicAdd(o + 1, v.y);
  atomicAdd(o + 2, v.z);
  atomicAdd(o + 3, v.w);
}

// ------- fused: t = AGG*inv_din + b ; ELU ; * inv_dout ; -> bf16 (layer-2 input) -------
__global__ void k_elu_cvt(const float* __restrict__ AGG, const float* __restrict__ inv_din,
                          const float* __restrict__ inv_dout, const float* __restrict__ b,
                          __bf16* __restrict__ Y, int F, long total4) {
  long idx = blockIdx.x * (long)blockDim.x + threadIdx.x;
  if (idx >= total4) return;
  int row = (int)(idx / (F / 4));
  int c   = (int)(idx % (F / 4)) * 4;
  float4 v = *(const float4*)&AGG[(long)row * F + c];
  float di = inv_din[row], dd = inv_dout[row];
  float t0 = v.x * di + b[c + 0];
  float t1 = v.y * di + b[c + 1];
  float t2 = v.z * di + b[c + 2];
  float t3 = v.w * di + b[c + 3];
  t0 = t0 > 0.f ? t0 : expm1f(t0);
  t1 = t1 > 0.f ? t1 : expm1f(t1);
  t2 = t2 > 0.f ? t2 : expm1f(t2);
  t3 = t3 > 0.f ? t3 : expm1f(t3);
  v4bf y;
  y[0] = (__bf16)(t0 * dd); y[1] = (__bf16)(t1 * dd);
  y[2] = (__bf16)(t2 * dd); y[3] = (__bf16)(t3 * dd);
  *(v4bf*)&Y[(long)row * F + c] = y;
}

// ---------------- out = out * inv_din + b2 (in place on d_out) ----------------
__global__ void k_final(float* __restrict__ out, const float* __restrict__ inv_din,
                        const float* __restrict__ b, int F, long total4) {
  long idx = blockIdx.x * (long)blockDim.x + threadIdx.x;
  if (idx >= total4) return;
  int row = (int)(idx / (F / 4));
  int c   = (int)(idx % (F / 4)) * 4;
  float4 v = *(float4*)&out[(long)row * F + c];
  float di = inv_din[row];
  v.x = v.x * di + b[c + 0];
  v.y = v.y * di + b[c + 1];
  v.z = v.z * di + b[c + 2];
  v.w = v.w * di + b[c + 3];
  *(float4*)&out[(long)row * F + c] = v;
}

extern "C" void kernel_launch(void* const* d_in, const int* in_sizes, int n_in,
                              void* d_out, int out_size, void* d_ws, size_t ws_size,
                              hipStream_t stream) {
  const float*     h   = (const float*)d_in[0];
  const float*     W1  = (const float*)d_in[1];
  const float*     b1  = (const float*)d_in[2];
  const float*     W2  = (const float*)d_in[3];
  const float*     b2  = (const float*)d_in[4];
  const long long* src = (const long long*)d_in[5];
  const long long* dst = (const long long*)d_in[6];
  const int N = in_sizes[0] / IN_DIM;   // 65536
  const int E = in_sizes[5];            // 524288

  // ---- workspace carve-up (all offsets 16B-aligned) ----
  char* ws = (char*)d_ws;
  float*  inv_dout = (float*)ws;                                  // N
  float*  inv_din  = inv_dout + N;                                // N
  __bf16* W1T      = (__bf16*)(inv_din + N);                      // HID*IN
  __bf16* W2T      = W1T + (size_t)HID_DIM * IN_DIM;              // OUT*HID
  __bf16* Hs       = W2T + (size_t)OUT_DIM * HID_DIM;             // N*IN bf16
  float*  X1       = (float*)(Hs + (size_t)N * IN_DIM);           // N*HID f32
  float*  AGG1     = X1 + (size_t)N * HID_DIM;                    // N*HID f32
  __bf16* H1s      = Hs;   // reuse (Hs dead after GEMM1)
  float*  X2       = X1;   // reuse (X1 dead after scatter1)
  float*  out      = (float*)d_out;

  const int T = 256;

  // degrees -> inv sqrt
  hipMemsetAsync(inv_dout, 0, sizeof(float) * 2 * (size_t)N, stream);
  k_deg<<<(E + T - 1) / T, T, 0, stream>>>(src, dst, inv_dout, inv_din, E);
  k_rsqrt<<<(2 * N + T - 1) / T, T, 0, stream>>>(inv_dout, 2 * N);

  // Hs = bf16(h * inv_dout) ; W^T bf16
  { long t8 = (long)N * IN_DIM / 8;
    k_scale_cvt<<<(int)((t8 + T - 1) / T), T, 0, stream>>>(h, inv_dout, Hs, IN_DIM, t8); }
  k_cvt_T<<<(IN_DIM * HID_DIM + T - 1) / T, T, 0, stream>>>(W1, W1T, IN_DIM, HID_DIM);
  k_cvt_T<<<(HID_DIM * OUT_DIM + T - 1) / T, T, 0, stream>>>(W2, W2T, HID_DIM, OUT_DIM);

  // Layer 1 GEMM: X1 = Hs @ W1
  { dim3 g(HID_DIM / 256, N / 64);
    gemm_bf16_wmma<<<g, T, 0, stream>>>(Hs, W1T, X1, N, HID_DIM, IN_DIM); }

  // Layer 1 aggregation
  hipMemsetAsync(AGG1, 0, sizeof(float) * (size_t)N * HID_DIM, stream);
  { long tot = (long)E * (HID_DIM / 4);
    k_scatter4<<<(int)((tot + T - 1) / T), T, 0, stream>>>(src, dst, X1, AGG1, HID_DIM / 4, tot); }

  // norm + bias + ELU + layer-2 pre-scale -> bf16
  { long t4 = (long)N * HID_DIM / 4;
    k_elu_cvt<<<(int)((t4 + T - 1) / T), T, 0, stream>>>(AGG1, inv_din, inv_dout, b1, H1s, HID_DIM, t4); }

  // Layer 2 GEMM: X2 = H1s @ W2
  { dim3 g(OUT_DIM / 256, N / 64);
    gemm_bf16_wmma<<<g, T, 0, stream>>>(H1s, W2T, X2, N, OUT_DIM, HID_DIM); }

  // Layer 2 aggregation directly into d_out
  hipMemsetAsync(out, 0, sizeof(float) * (size_t)N * OUT_DIM, stream);
  { long tot = (long)E * (OUT_DIM / 4);
    k_scatter4<<<(int)((tot + T - 1) / T), T, 0, stream>>>(src, dst, X2, out, OUT_DIM / 4, tot); }

  // finalize: out = out * inv_din + b2
  { long t4 = (long)N * OUT_DIM / 4;
    k_final<<<(int)((t4 + T - 1) / T), T, 0, stream>>>(out, inv_din, b2, OUT_DIM, t4); }
}
